// PointConvFormerEncoder_81475529605616
// MI455X (gfx1250) — matrix operand order repaired and checked by reading, not compiled, not run on hardware
//
#include <hip/hip_runtime.h>
#include <hip/hip_bf16.h>

// ---------------------------------------------------------------------------
// PointConvFormer encoder for MI455X (gfx1250, wave32, WMMA).
// Heavy GEMMs (agg @ Wout) run on v_wmma_f32_16x16x32_f16 (f16 in, f32 acc),
// one wave per 16x64 output tile: 1 A-fragment feeds 4 WMMAs per K-step.
// ---------------------------------------------------------------------------

typedef _Float16 half_t;
typedef __attribute__((ext_vector_type(16))) _Float16 v16h;
typedef __attribute__((ext_vector_type(8)))  _Float16 v8h;
typedef __attribute__((ext_vector_type(8)))  float    v8f;

#define KNN 16
#define MID 16
#define HEADS 8

// -------------------------------- subsample --------------------------------
__global__ void subsample_kernel(const float* __restrict__ src, float* __restrict__ dst,
                                 int nPerOut, int nPerIn, int nBatch) {
    int i = blockIdx.x * blockDim.x + threadIdx.x;
    int tot = nBatch * nPerOut;
    if (i >= tot) return;
    int b = i / nPerOut, p = i - b * nPerOut;
    int si = b * nPerIn + p * 4;           // STRIDE = 4
    dst[i * 3 + 0] = src[si * 3 + 0];
    dst[i * 3 + 1] = src[si * 3 + 1];
    dst[i * 3 + 2] = src[si * 3 + 2];
}

// ---------------------------------- KNN ------------------------------------
// One thread per query point: brute-force scan of its batch's candidates,
// register-resident sorted top-16 (bubble-restore after tail replacement).
__global__ void knn16_kernel(const float* __restrict__ qp, const float* __restrict__ kp,
                             int nqPer, int nkPer, int* __restrict__ edges, int nTotQ) {
    int qi = blockIdx.x * blockDim.x + threadIdx.x;
    if (qi >= nTotQ) return;
    int b = qi / nqPer;
    int base = b * nkPer;
    float qx = qp[qi * 3 + 0], qy = qp[qi * 3 + 1], qz = qp[qi * 3 + 2];
    float bd[KNN]; int bi[KNN];
#pragma unroll
    for (int s = 0; s < KNN; ++s) { bd[s] = 3.4e38f; bi[s] = base; }
    for (int j = 0; j < nkPer; ++j) {
        float dx = kp[(base + j) * 3 + 0] - qx;
        float dy = kp[(base + j) * 3 + 1] - qy;
        float dz = kp[(base + j) * 3 + 2] - qz;
        float d = dx * dx + dy * dy + dz * dz;
        if (d < bd[KNN - 1]) {
            bd[KNN - 1] = d; bi[KNN - 1] = base + j;
#pragma unroll
            for (int s = KNN - 1; s > 0; --s) {
                if (bd[s] < bd[s - 1]) {
                    float td = bd[s]; bd[s] = bd[s - 1]; bd[s - 1] = td;
                    int   ti = bi[s]; bi[s] = bi[s - 1]; bi[s - 1] = ti;
                }
            }
        }
    }
#pragma unroll
    for (int s = 0; s < KNN; ++s) edges[qi * KNN + s] = bi[s];
}

// ------------------------- weight prep (f32 -> f16^T) ----------------------
// Wt[n*Kp + k] = (f16) Wout[k*Cout + n], zero-padded for k in [Kdim, Kp).
__global__ void prep_w_kernel(const float* __restrict__ W, _Float16* __restrict__ Wt,
                              int Kdim, int Kp, int Cout) {
    int idx = blockIdx.x * blockDim.x + threadIdx.x;
    int total = Cout * Kp;
    if (idx >= total) return;
    int n = idx / Kp, k = idx - n * Kp;
    Wt[idx] = (k < Kdim) ? (_Float16)W[(size_t)k * Cout + n] : (_Float16)0.f;
}

// ------------------- gather + pos-MLP + attention + einsum -----------------
// One 256-thread block per query point. Produces agg row (f16, padded to Kp).
__global__ void __launch_bounds__(256) pcf_gather_kernel(
    const float* __restrict__ ptsQ, const float* __restrict__ ptsK,
    const float* __restrict__ feats, const int* __restrict__ edges,
    const float* __restrict__ Wp1, const float* __restrict__ Wp2,
    const float* __restrict__ Wa, _Float16* __restrict__ agg,
    int Cin, int Kp) {
    extern __shared__ float smem[];
    float* s_nbr = smem;                  // 16 * Cin
    float* s_w   = smem + 16 * Cin;       // 16 * 16
    float* s_at  = s_w + 256;             // 16 * 8
    __shared__ int s_e[KNN];

    const int q = blockIdx.x;
    const int tid = threadIdx.x;
    if (tid < KNN) s_e[tid] = edges[q * KNN + tid];
    __syncthreads();

    // Gather neighbor features into LDS (K x Cin).
    for (int idx = tid; idx < KNN * Cin; idx += 256) {
        int k = idx / Cin, c = idx - k * Cin;
        s_nbr[idx] = feats[(size_t)s_e[k] * Cin + c];
    }

    // Positional MLP: w[k][m] = (relu(rel @ Wp1) @ Wp2)[m], one thread each.
    {
        int k = tid >> 4, m = tid & 15;
        float rx = ptsK[s_e[k] * 3 + 0] - ptsQ[q * 3 + 0];
        float ry = ptsK[s_e[k] * 3 + 1] - ptsQ[q * 3 + 1];
        float rz = ptsK[s_e[k] * 3 + 2] - ptsQ[q * 3 + 2];
        float acc = 0.f;
#pragma unroll
        for (int j = 0; j < MID; ++j) {
            float h = rx * Wp1[j] + ry * Wp1[16 + j] + rz * Wp1[32 + j];
            h = h > 0.f ? h : 0.f;
            acc += h * Wp2[j * 16 + m];
        }
        s_w[tid] = acc;
    }
    __syncthreads();

    // Attention logits: (K x HEADS) = nbr @ Wa.
    if (tid < KNN * HEADS) {
        int k = tid >> 3, h = tid & 7;
        float acc = 0.f;
        for (int c = 0; c < Cin; ++c) acc += s_nbr[k * Cin + c] * Wa[c * HEADS + h];
        s_at[k * HEADS + h] = acc;
    }
    __syncthreads();

    // Softmax over K per head.
    if (tid < HEADS) {
        float mx = -3.4e38f;
#pragma unroll
        for (int k = 0; k < KNN; ++k) { float v = s_at[k * HEADS + tid]; mx = v > mx ? v : mx; }
        float ex[KNN]; float sum = 0.f;
#pragma unroll
        for (int k = 0; k < KNN; ++k) { ex[k] = expf(s_at[k * HEADS + tid] - mx); sum += ex[k]; }
        float inv = 1.f / sum;
#pragma unroll
        for (int k = 0; k < KNN; ++k) s_at[k * HEADS + tid] = ex[k] * inv;
    }
    __syncthreads();

    // w[k][m] *= attn[k][m/2]   (MID/HEADS == 2)
    {
        int k = tid >> 4, m = tid & 15;
        s_w[tid] *= s_at[k * HEADS + (m >> 1)];
    }
    __syncthreads();

    // agg[q][c*16+m] = sum_k nbr[k][c] * w[k][m]; zero padding to Kp. f16 out.
    const int Kdim = Cin * MID;
    for (int oc = tid; oc < Kp; oc += 256) {
        float v = 0.f;
        if (oc < Kdim) {
            int c = oc >> 4, m = oc & 15;
#pragma unroll
            for (int k = 0; k < KNN; ++k) v += s_nbr[k * Cin + c] * s_w[k * 16 + m];
        }
        agg[(size_t)q * Kp + oc] = (_Float16)v;
    }
}

// ------------------------ WMMA GEMM + bias + ReLU --------------------------
// out(Nq x Cout) = relu(A(Nq x Kp, f16) @ Wt^T + bias), Wt is Cout x Kp f16.
// One wave per 16(M) x 64(N) tile: per 32-wide K-step, one A fragment feeds
// 4 v_wmma_f32_16x16x32_f16 against 4 B fragments (6 loads : 4 WMMAs).
__global__ void __launch_bounds__(256) gemm_bias_relu_wmma(
    const _Float16* __restrict__ A, const _Float16* __restrict__ Bt,
    const float* __restrict__ bias, float* __restrict__ out,
    int Nq, int Kp, int Cout) {
    const int lane = threadIdx.x & 31;
    const int wid  = blockIdx.x * (blockDim.x >> 5) + (threadIdx.x >> 5);
    const int tilesN = Cout >> 6;                // 64-wide N tiles
    const int tilesM = Nq >> 4;
    if (wid >= tilesM * tilesN) return;          // wave-uniform exit (EXEC stays full)
    const int tm = wid / tilesN, tn = wid - tm * tilesN;
    const int hf = lane >> 4;                    // lane half (0: lanes 0-15, 1: 16-31)
    const int r  = lane & 15;

    const _Float16* arow = A  + (size_t)(tm * 16 + r) * Kp;        // A row M=r
    const _Float16* brow = Bt + (size_t)(tn * 64 + r) * Kp;        // B cols, sub-tile j at +j*16*Kp

    v8f acc[4] = {v8f{}, v8f{}, v8f{}, v8f{}};
    for (int k = 0; k < Kp; k += 32) {
        if (k + 256 < Kp) __builtin_prefetch(arow + k + 256, 0, 1);   // stream A ahead
        // A fragment: elems 0-15 -> K = k + 16*hf + [0..15]
        v8h a0 = *(const v8h*)(arow + k + 16 * hf);
        v8h a1 = *(const v8h*)(arow + k + 16 * hf + 8);
        v16h av;
#pragma unroll
        for (int i = 0; i < 8; ++i) { av[i] = a0[i]; av[i + 8] = a1[i]; }
        // B fragments: lanes 0-15 hold K = k+[0..15], lanes 16-31 hold K = k+16+[0..15]
        v16h bv[4];
#pragma unroll
        for (int j = 0; j < 4; ++j)
            bv[j] = *(const v16h*)(brow + (size_t)j * 16 * Kp + k + 16 * hf);
#pragma unroll
        for (int j = 0; j < 4; ++j)
            acc[j] = __builtin_amdgcn_wmma_f32_16x16x32_f16(false, av, false, bv[j],
                                                            (short)0, acc[j], false, false);
    }
#pragma unroll
    for (int j = 0; j < 4; ++j) {
        const int ncol = tn * 64 + j * 16 + r;
        const float bb = bias[ncol];
#pragma unroll
        for (int v = 0; v < 8; ++v) {
            int m = tm * 16 + v + 8 * hf;        // C/D layout: VGPR v, lane-half -> M
            float val = acc[j][v] + bb;
            out[(size_t)m * Cout + ncol] = val > 0.f ? val : 0.f;
        }
    }
}

// --------------------------- pooling + final linear ------------------------
__global__ void pool_kernel(const float* __restrict__ x, float* __restrict__ pooled,
                            int C, int nFinalPer, int nBatch) {
    int i = blockIdx.x * blockDim.x + threadIdx.x;
    if (i >= nBatch * C) return;
    int b = i / C, c = i - b * C;
    float s = 0.f;
    for (int p = 0; p < nFinalPer; ++p) s += x[(size_t)(b * nFinalPer + p) * C + c];
    pooled[i] = s / (float)nFinalPer;
}

__global__ void final_linear_kernel(const float* __restrict__ pooled,
                                    const float* __restrict__ W,
                                    const float* __restrict__ b,
                                    float* __restrict__ out, int C, int O, int nBatch) {
    int i = blockIdx.x * blockDim.x + threadIdx.x;
    if (i >= nBatch * O) return;
    int bb = i / O, o = i - bb * O;
    float s = b[o];
    for (int c = 0; c < C; ++c) s += pooled[bb * C + c] * W[c * O + o];
    out[i] = s;
}

// ------------------------------- host driver -------------------------------
static void run_layer(hipStream_t s, const float* ptsQ, const float* ptsK,
                      const float* feats, int Cin, const int* edges,
                      const float* Wp1, const float* Wp2, const float* Wa,
                      const float* Wout, const float* bias, int Cout,
                      float* out, int Nq, _Float16* agg, _Float16* Wt) {
    const int Kdim = Cin * MID;
    const int Kp = (Kdim + 31) & ~31;
    {
        int total = Cout * Kp;
        prep_w_kernel<<<(total + 255) / 256, 256, 0, s>>>(Wout, Wt, Kdim, Kp, Cout);
    }
    {
        size_t smem = (size_t)(16 * Cin + 256 + 128) * sizeof(float);
        pcf_gather_kernel<<<Nq, 256, smem, s>>>(ptsQ, ptsK, feats, edges,
                                                Wp1, Wp2, Wa, agg, Cin, Kp);
    }
    {
        int tiles = (Nq >> 4) * (Cout >> 6);     // 16x64 tiles per wave
        gemm_bias_relu_wmma<<<(tiles + 7) / 8, 256, 0, s>>>(agg, Wt, bias, out, Nq, Kp, Cout);
    }
}

extern "C" void kernel_launch(void* const* d_in, const int* in_sizes, int n_in,
                              void* d_out, int out_size, void* d_ws, size_t ws_size,
                              hipStream_t stream) {
    (void)in_sizes; (void)n_in; (void)out_size; (void)ws_size;

    const int Bb = 8;
    const int nPer[5] = {4096, 1024, 256, 64, 16};
    const int nTot[5] = {32768, 8192, 2048, 512, 128};
    const int FEAT[5] = {64, 128, 256, 512, 512};

    const float* points   = (const float*)d_in[0];
    const float* features = (const float*)d_in[1];
    // d_in[2] = batch (unused: layout is implicit)
    // params_self[i] at 3+5i, params_fwd[j] at 28+5j, each (Wp1,Wp2,Wa,Wout,b)
    const float* lin_W = (const float*)d_in[48];
    const float* lin_b = (const float*)d_in[49];

    // ---- carve workspace ----
    char* wp = (char*)d_ws;
    auto carve = [&](size_t bytes) -> void* {
        void* r = (void*)wp;
        wp += (bytes + 255) & ~(size_t)255;
        return r;
    };
    float* lvl[5];
    lvl[0] = (float*)points;
    for (int l = 1; l < 5; ++l) lvl[l] = (float*)carve((size_t)nTot[l] * 3 * sizeof(float));
    int* eS[5]; int* eF[4];
    for (int l = 0; l < 5; ++l) eS[l] = (int*)carve((size_t)nTot[l] * KNN * sizeof(int));
    for (int l = 0; l < 4; ++l) eF[l] = (int*)carve((size_t)nTot[l + 1] * KNN * sizeof(int));
    float* xA = (float*)carve((size_t)32768 * 64 * sizeof(float));   // 8 MB (max A-side act)
    float* xB = (float*)carve((size_t)8192 * 128 * sizeof(float));   // 4 MB (max B-side act)
    _Float16* agg = (_Float16*)carve((size_t)8192 * 2048 * sizeof(_Float16)); // 32 MB max
    _Float16* Wt  = (_Float16*)carve((size_t)512 * 8192 * sizeof(_Float16)); // 8 MB max
    float* pooled = (float*)carve((size_t)Bb * 512 * sizeof(float));

    // ---- subsample levels (stride 4 within each batch) ----
    for (int l = 1; l < 5; ++l) {
        int tot = nTot[l];
        subsample_kernel<<<(tot + 255) / 256, 256, 0, stream>>>(lvl[l - 1], lvl[l],
                                                                nPer[l], nPer[l - 1], Bb);
    }

    // ---- KNN graphs ----
    for (int l = 0; l < 5; ++l)
        knn16_kernel<<<(nTot[l] + 255) / 256, 256, 0, stream>>>(lvl[l], lvl[l],
                                                                nPer[l], nPer[l], eS[l], nTot[l]);
    for (int l = 0; l < 4; ++l)
        knn16_kernel<<<(nTot[l + 1] + 255) / 256, 256, 0, stream>>>(lvl[l + 1], lvl[l],
                                                                    nPer[l + 1], nPer[l], eF[l], nTot[l + 1]);

    auto P = [&](int base, int j) -> const float* { return (const float*)d_in[base + j]; };
    // params_self[i] base = 3 + 5*i ; params_fwd[j] base = 28 + 5*j

    // ---- layer pipeline ----
    // self0: (3 -> 64) at level 0
    {
        int b0 = 3;
        run_layer(stream, lvl[0], lvl[0], features, 3, eS[0],
                  P(b0, 0), P(b0, 1), P(b0, 2), P(b0, 3), P(b0, 4), FEAT[0],
                  xA, nTot[0], agg, Wt);
    }
    const float* curA = xA;  // features at current level (in xA after self layers)
    for (int i = 0; i < 4; ++i) {
        int bf = 28 + 5 * i;           // fwd layer i: FEAT[i] -> FEAT[i+1]
        run_layer(stream, lvl[i + 1], lvl[i], curA, FEAT[i], eF[i],
                  P(bf, 0), P(bf, 1), P(bf, 2), P(bf, 3), P(bf, 4), FEAT[i + 1],
                  xB, nTot[i + 1], agg, Wt);
        int bs = 3 + 5 * (i + 1);      // self layer at level i+1: FEAT[i+1] -> FEAT[i+1]
        run_layer(stream, lvl[i + 1], lvl[i + 1], xB, FEAT[i + 1], eS[i + 1],
                  P(bs, 0), P(bs, 1), P(bs, 2), P(bs, 3), P(bs, 4), FEAT[i + 1],
                  xA, nTot[i + 1], agg, Wt);
        curA = xA;
    }

    // ---- mean pool over the 16 final points per batch + final linear ----
    pool_kernel<<<(Bb * 512 + 255) / 256, 256, 0, stream>>>(xA, pooled, 512, nPer[4], Bb);
    final_linear_kernel<<<(Bb * 256 + 255) / 256, 256, 0, stream>>>(pooled, lin_W, lin_b,
                                                                    (float*)d_out, 512, 256, Bb);
}
